// CustomScaledDotProductAttention_45603962748940
// MI455X (gfx1250) — compile-verified, matrix operand order checked by reading, not run on hardware
//
#include <hip/hip_runtime.h>

// ---------------------------------------------------------------------------
// MI455X (gfx1250) multi-head attention forward.
//   B=4, NQ=NK=1024, F=1024, H=16, dh=64.
// All matmuls via v_wmma_f32_16x16x32_bf16 (wave32); K/mask tiles staged into
// LDS by the Tensor Data Mover (tensor_load_to_lds + s_wait_tensorcnt).
// d_out = [results f32: 4*1024*1024] ++ [distribution f32: 4*1024*16*1024]
// d_ws  = q,k,v,ctx as bf16 (8 MB each, 32 MB total)
// ---------------------------------------------------------------------------

typedef __attribute__((ext_vector_type(16))) __bf16 v16bf;
typedef __attribute__((ext_vector_type(8)))  __bf16 v8bf;
typedef __attribute__((ext_vector_type(4)))  __bf16 v4bf;
typedef __attribute__((ext_vector_type(8)))  float  v8f;
typedef __attribute__((ext_vector_type(4)))  unsigned int u32x4;
typedef __attribute__((ext_vector_type(8)))  int i32x8;
typedef __attribute__((ext_vector_type(4)))  int i32x4;

#define NF    1024
#define HEADS 16
#define DH    64
#define NB    4
#define NQK   1024

#define WMMA(a,b,c) __builtin_amdgcn_wmma_f32_16x16x32_bf16(false,(a),false,(b),(short)0,(c),false,false)

#if __has_builtin(__builtin_amdgcn_tensor_load_to_lds) && __has_builtin(__builtin_amdgcn_s_wait_tensorcnt)
#define USE_TDM 1
#else
#define USE_TDM 0
#endif

static __device__ __forceinline__ v16bf mkfrag(v8bf lo, v8bf hi) {
  return __builtin_shufflevector(lo, hi, 0,1,2,3,4,5,6,7,8,9,10,11,12,13,14,15);
}
static __device__ __forceinline__ v8f vzero8() {
  v8f z = {0.f,0.f,0.f,0.f,0.f,0.f,0.f,0.f};
  return z;
}

#if USE_TDM
// Issue one TDM 2-D tile load: global (row-major, stride0 elems/row) -> LDS,
// with hardware row padding so LDS rows land on our padded strides.
// dsz_code: 0=1B, 1=2B elements. pad_int_code: 2^(n+1) DWORDs between pads.
// pad_amt_code: (pad DWORDs - 1).
// This toolchain's builtin takes 6 args: (g0, g1, g2, g3, g4, cpol);
// groups 2/3/4 are unused for a 2-D tile -> zero-filled.
static __device__ __forceinline__ void tdm_load_2d(
    unsigned long long gaddr, unsigned lds_off, unsigned dsz_code,
    unsigned tensor_d0, unsigned tensor_d1, unsigned tile_d0, unsigned tile_d1,
    unsigned long long stride0, unsigned pad_int_code, unsigned pad_amt_code)
{
  union { unsigned long long q[2]; u32x4 v; } g0;
  union { unsigned long long q[4]; i32x8 v; } g1;
  // group0: count=1 | lds_addr[63:32] ; global_addr[120:64] | type=2 [127:126]
  g0.q[0] = 1ull | ((unsigned long long)lds_off << 32);
  g0.q[1] = (gaddr & ((1ull << 57) - 1ull)) | (2ull << 62);
  // group1 bitfields (see ISA 8.4)
  unsigned long long dw0 = (unsigned long long)((dsz_code << 16) | (1u << 20) |
                                                (pad_int_code << 22) | (pad_amt_code << 25));
  g1.q[0] = dw0 | (((unsigned long long)tensor_d0 & 0xFFFFull) << 48);        // tensor_dim0[15:0]
  g1.q[1] = (((unsigned long long)tensor_d0 >> 16) & 0xFFFFull)               // tensor_dim0[31:16]
          | (((unsigned long long)tensor_d1 & 0xFFFFFFFFull) << 16)           // tensor_dim1
          | (((unsigned long long)tile_d0 & 0xFFFFull) << 48);                // tile_dim0
  g1.q[2] = ((unsigned long long)tile_d1 & 0xFFFFull)                         // tile_dim1 (tile_dim2=0)
          | ((stride0 & 0xFFFFFFFFull) << 32);                                // stride0[31:0]
  g1.q[3] = (stride0 >> 32) & 0xFFFFull;                                      // stride0[47:32]; stride1=0
  i32x4 z4; z4[0] = z4[1] = z4[2] = z4[3] = 0;
  i32x8 z8;
  #pragma unroll
  for (int i = 0; i < 8; ++i) z8[i] = 0;
  __builtin_amdgcn_tensor_load_to_lds(g0.v, g1.v, z4, z4, z8, 0);
}
#endif

// ---------------------------------------------------------------------------
// Tiled GEMM:  C[M=4096,N=1024] = A[M,K=1024] * W[N,K]^T + bias, then either
//   QKV=true : out bf16 at [b,h,n,dh];  QKV=false: out f32 row-major.
// scale multiplies (acc+bias) — folds 1/sqrt(dh) into q.
// 256 threads (8 waves), tile 128x128, K-step 32. f32->bf16 at LDS stage.
// ---------------------------------------------------------------------------
static constexpr int Kdim = 1024;
static constexpr int BM = 128, BN = 128, BK = 32, SK = 40;

template<typename AT, bool QKV>
__global__ void __launch_bounds__(256)
gemm_kernel(const AT* __restrict__ A, const float* __restrict__ W,
            const float* __restrict__ bias, void* __restrict__ outp, float scale)
{
  __shared__ alignas(16) __bf16 As[BM * SK];
  __shared__ alignas(16) __bf16 Ws[BN * SK];

  const int tid  = threadIdx.x;
  const int wave = tid >> 5, lane = tid & 31, lr = lane & 15, lg = lane >> 4;
  const int wm = wave & 1, wn = wave >> 1;
  const int m0 = blockIdx.y * BM, n0 = blockIdx.x * BN;

  v8f acc[4][2];
  #pragma unroll
  for (int i = 0; i < 4; ++i)
    #pragma unroll
    for (int j = 0; j < 2; ++j) acc[i][j] = vzero8();

  for (int k0 = 0; k0 < Kdim; k0 += BK) {
    __syncthreads();
    #pragma unroll
    for (int it = 0; it < 4; ++it) {
      const int idx = it * 256 + tid;
      const int row = idx >> 3, c4 = (idx & 7) * 4;
      v4bf av;
      if constexpr (sizeof(AT) == 4) {
        const float4 t = *(const float4*)(A + (size_t)(m0 + row) * Kdim + k0 + c4);
        av[0] = (__bf16)t.x; av[1] = (__bf16)t.y; av[2] = (__bf16)t.z; av[3] = (__bf16)t.w;
      } else {
        av = *(const v4bf*)(A + (size_t)(m0 + row) * Kdim + k0 + c4);
      }
      *(v4bf*)&As[row * SK + c4] = av;

      const float4 w = *(const float4*)(W + (size_t)(n0 + row) * Kdim + k0 + c4);
      v4bf wv;
      wv[0] = (__bf16)w.x; wv[1] = (__bf16)w.y; wv[2] = (__bf16)w.z; wv[3] = (__bf16)w.w;
      *(v4bf*)&Ws[row * SK + c4] = wv;

      if (k0 + BK < Kdim) {  // global_prefetch_b8 of next K-tile
        __builtin_prefetch(A + (size_t)(m0 + row) * Kdim + (k0 + BK) + c4);
        __builtin_prefetch(W + (size_t)(n0 + row) * Kdim + (k0 + BK) + c4);
      }
    }
    __syncthreads();

    v16bf af[4], bf[2];
    #pragma unroll
    for (int i = 0; i < 4; ++i) {
      const __bf16* p = &As[(wm * 64 + i * 16 + lr) * SK + lg * 8];
      af[i] = mkfrag(*(const v8bf*)p, *(const v8bf*)(p + 16));
    }
    #pragma unroll
    for (int j = 0; j < 2; ++j) {
      const __bf16* p = &Ws[(wn * 32 + j * 16 + lr) * SK + lg * 8];
      bf[j] = mkfrag(*(const v8bf*)p, *(const v8bf*)(p + 16));
    }
    #pragma unroll
    for (int i = 0; i < 4; ++i)
      #pragma unroll
      for (int j = 0; j < 2; ++j)
        acc[i][j] = WMMA(af[i], bf[j], acc[i][j]);
  }

  #pragma unroll
  for (int i = 0; i < 4; ++i)
    #pragma unroll
    for (int j = 0; j < 2; ++j)
      #pragma unroll
      for (int r = 0; r < 8; ++r) {
        const int grow = m0 + wm * 64 + i * 16 + r + lg * 8;
        const int gcol = n0 + wn * 32 + j * 16 + lr;
        const float v = (acc[i][j][r] + bias[gcol]) * scale;
        if constexpr (QKV) {
          const int b = grow >> 10, nq = grow & 1023, h = gcol >> 6, d = gcol & 63;
          ((__bf16*)outp)[(((size_t)(b * HEADS + h) * NQK) + nq) * DH + d] = (__bf16)v;
        } else {
          ((float*)outp)[(size_t)grow * NF + gcol] = v;
        }
      }
}

// ---------------------------------------------------------------------------
// Attention: block = one (b,h) x 64 query rows; 4 waves, each owns 16 full
// rows (softmax stats never cross waves). Two-pass online softmax; S is
// recomputed in pass 2 (cheap on WMMA) so the 268MB distribution is written
// exactly once and never re-read. K + mask tiles staged by the TDM.
// ---------------------------------------------------------------------------
__global__ void __launch_bounds__(128)
attn_kernel(const __bf16* __restrict__ qb, const __bf16* __restrict__ kb,
            const __bf16* __restrict__ vb, const unsigned char* __restrict__ mask,
            float* __restrict__ dist, __bf16* __restrict__ ctx)
{
  constexpr int SV = 72;                       // padded row: 144B (36 DWORDs)
  __shared__ alignas(16) __bf16 ks [64 * SV];  // K tile [key][d]
  __shared__ alignas(16) __bf16 vts[64 * SV];  // V tile transposed [d][key]
  __shared__ alignas(16) __bf16 pbuf[4 * 16 * SV];
  __shared__ alignas(16) unsigned char msk[64 * 72]; // mask tile [qrow][kcol], 72B rows

  const int tid  = threadIdx.x;
  const int wave = tid >> 5, lane = tid & 31, lr = lane & 15, lg = lane >> 4;
  const int bh = blockIdx.x, b = bh >> 4, h = bh & 15;
  const int q0 = blockIdx.y * 64;
  const int row0 = q0 + wave * 16;

  const __bf16* qp = qb + (size_t)bh * NQK * DH;
  const __bf16* kp = kb + (size_t)bh * NQK * DH;
  const __bf16* vp = vb + (size_t)bh * NQK * DH;
  const unsigned char* mp = mask + ((size_t)b * NQK + q0) * NQK; // row q0, col 0

  // q fragments (16 rows x 64 features = 2 K-steps), resident all kernel
  v16bf aq[2];
  #pragma unroll
  for (int s = 0; s < 2; ++s) {
    const __bf16* p = qp + (size_t)(row0 + lr) * DH + s * 32 + lg * 8;
    aq[s] = mkfrag(*(const v8bf*)p, *(const v8bf*)(p + 16));
  }

  float m[8], l[8];
  #pragma unroll
  for (int r = 0; r < 8; ++r) { m[r] = -__builtin_inff(); l[r] = 0.f; }

  // ---------------- pass 1: online softmax statistics ----------------
  for (int kt = 0; kt < 16; ++kt) {
    __syncthreads();
#if USE_TDM
    if (wave == 0) {
      // K tile: 64 keys x 64 bf16, contiguous rows; pad 32dw data + 4dw -> 144B rows
      tdm_load_2d((unsigned long long)(size_t)(kp + (size_t)kt * 64 * DH),
                  (unsigned)(size_t)ks, 1, 64, 1024, 64, 64, 64ull, 4, 3);
      // mask tile: 64 q-rows x 64 bytes, row stride 1024; pad 16dw + 2dw -> 72B rows
      tdm_load_2d((unsigned long long)(size_t)(mp + kt * 64),
                  (unsigned)(size_t)msk, 0, 1024, 1024, 64, 64, 1024ull, 3, 1);
      __builtin_amdgcn_s_wait_tensorcnt(0);
    }
#else
    #pragma unroll
    for (int it = 0; it < 4; ++it) {
      const int idx = it * 128 + tid;
      const int row = idx >> 3, c8 = (idx & 7) * 8;
      *(v8bf*)&ks[row * SV + c8] = *(const v8bf*)(kp + (size_t)(kt * 64 + row) * DH + c8);
    }
    #pragma unroll
    for (int it = 0; it < 2; ++it) {
      const int idx = it * 128 + tid;
      const int row = idx >> 1, c = (idx & 1) * 32;
      *(uint4*)&msk[row * 72 + c] = *(const uint4*)(mp + (size_t)row * NQK + kt * 64 + c);
      *(uint4*)&msk[row * 72 + c + 16] = *(const uint4*)(mp + (size_t)row * NQK + kt * 64 + c + 16);
    }
#endif
    __syncthreads();

    v8f sac[4];
    #pragma unroll
    for (int j = 0; j < 4; ++j) {
      sac[j] = vzero8();
      #pragma unroll
      for (int s = 0; s < 2; ++s) {
        const __bf16* p = &ks[(j * 16 + lr) * SV + s * 32 + lg * 8];
        sac[j] = WMMA(aq[s], mkfrag(*(const v8bf*)p, *(const v8bf*)(p + 16)), sac[j]);
      }
      #pragma unroll
      for (int r = 0; r < 8; ++r) {
        if (!msk[(wave * 16 + r + lg * 8) * 72 + j * 16 + lr]) sac[j][r] = -__builtin_inff();
      }
    }
    #pragma unroll
    for (int r = 0; r < 8; ++r) {
      float tm = fmaxf(fmaxf(sac[0][r], sac[1][r]), fmaxf(sac[2][r], sac[3][r]));
      #pragma unroll
      for (int off = 1; off < 16; off <<= 1) tm = fmaxf(tm, __shfl_xor(tm, off, 32));
      const float mn = fmaxf(m[r], tm);
      float ps = 0.f;
      #pragma unroll
      for (int j = 0; j < 4; ++j) ps += __expf(sac[j][r] - mn);
      #pragma unroll
      for (int off = 1; off < 16; off <<= 1) ps += __shfl_xor(ps, off, 32);
      l[r] = l[r] * __expf(m[r] - mn) + ps;
      m[r] = mn;
    }
  }

  // ---------------- pass 2: probabilities + context ----------------
  float rl[8];
  #pragma unroll
  for (int r = 0; r < 8; ++r) rl[r] = 1.f / l[r];

  v8f cacc[4];
  #pragma unroll
  for (int j = 0; j < 4; ++j) cacc[j] = vzero8();

  for (int kt = 0; kt < 16; ++kt) {
    __syncthreads();
#if USE_TDM
    if (wave == 0) {
      tdm_load_2d((unsigned long long)(size_t)(kp + (size_t)kt * 64 * DH),
                  (unsigned)(size_t)ks, 1, 64, 1024, 64, 64, 64ull, 4, 3);
      tdm_load_2d((unsigned long long)(size_t)(mp + kt * 64),
                  (unsigned)(size_t)msk, 0, 1024, 1024, 64, 64, 1024ull, 3, 1);
    }
#else
    #pragma unroll
    for (int it = 0; it < 4; ++it) {
      const int idx = it * 128 + tid;
      const int row = idx >> 3, c8 = (idx & 7) * 8;
      *(v8bf*)&ks[row * SV + c8] = *(const v8bf*)(kp + (size_t)(kt * 64 + row) * DH + c8);
    }
    #pragma unroll
    for (int it = 0; it < 2; ++it) {
      const int idx = it * 128 + tid;
      const int row = idx >> 1, c = (idx & 1) * 32;
      *(uint4*)&msk[row * 72 + c] = *(const uint4*)(mp + (size_t)row * NQK + kt * 64 + c);
      *(uint4*)&msk[row * 72 + c + 16] = *(const uint4*)(mp + (size_t)row * NQK + kt * 64 + c + 16);
    }
#endif
    // V tile transposed into LDS (TDM cannot transpose) — all waves help.
    #pragma unroll
    for (int it = 0; it < 4; ++it) {
      const int idx = it * 128 + tid;
      const int row = idx >> 3, c8 = (idx & 7) * 8;
      const v8bf vv = *(const v8bf*)(vp + (size_t)(kt * 64 + row) * DH + c8);
      #pragma unroll
      for (int u = 0; u < 8; ++u) vts[(c8 + u) * SV + row] = vv[u];
    }
#if USE_TDM
    if (wave == 0) __builtin_amdgcn_s_wait_tensorcnt(0);
#endif
    __syncthreads();

    #pragma unroll
    for (int j = 0; j < 4; ++j) {
      v8f sac = vzero8();
      #pragma unroll
      for (int s = 0; s < 2; ++s) {
        const __bf16* p = &ks[(j * 16 + lr) * SV + s * 32 + lg * 8];
        sac = WMMA(aq[s], mkfrag(*(const v8bf*)p, *(const v8bf*)(p + 16)), sac);
      }
      #pragma unroll
      for (int r = 0; r < 8; ++r) {
        const int qrow = row0 + r + lg * 8;
        const int kcol = kt * 64 + j * 16 + lr;
        float sv = msk[(wave * 16 + r + lg * 8) * 72 + j * 16 + lr] ? sac[r]
                                                                    : -__builtin_inff();
        const float p = __expf(sv - m[r]) * rl[r];
        dist[(((size_t)b * NQK + qrow) * HEADS + h) * NQK + kcol] = p;
        pbuf[(wave * 16 + r + lg * 8) * SV + j * 16 + lr] = (__bf16)p;
      }
    }
    __builtin_amdgcn_wave_barrier();  // keep LDS p-stores before the fragment loads

    v16bf pa[2];
    #pragma unroll
    for (int s = 0; s < 2; ++s) {
      const __bf16* pp = &pbuf[(wave * 16 + lr) * SV + s * 32 + lg * 8];
      pa[s] = mkfrag(*(const v8bf*)pp, *(const v8bf*)(pp + 16));
    }
    #pragma unroll
    for (int j2 = 0; j2 < 4; ++j2)
      #pragma unroll
      for (int s = 0; s < 2; ++s) {
        const __bf16* vv = &vts[(j2 * 16 + lr) * SV + s * 32 + lg * 8];
        cacc[j2] = WMMA(pa[s], mkfrag(*(const v8bf*)vv, *(const v8bf*)(vv + 16)), cacc[j2]);
      }
  }

  #pragma unroll
  for (int j2 = 0; j2 < 4; ++j2)
    #pragma unroll
    for (int r = 0; r < 8; ++r) {
      const int qrow = row0 + r + lg * 8;
      ctx[((size_t)b * NQK + qrow) * NF + h * DH + j2 * 16 + lr] = (__bf16)cacc[j2][r];
    }
}

// ---------------------------------------------------------------------------
extern "C" void kernel_launch(void* const* d_in, const int* in_sizes, int n_in,
                              void* d_out, int out_size, void* d_ws, size_t ws_size,
                              hipStream_t stream) {
  (void)in_sizes; (void)n_in; (void)out_size; (void)ws_size;
  const float* queries = (const float*)d_in[0];
  const float* keys    = (const float*)d_in[1];
  const float* values  = (const float*)d_in[2];
  const unsigned char* mask = (const unsigned char*)d_in[3];
  const float* Wq = (const float*)d_in[4];
  const float* bq = (const float*)d_in[5];
  const float* Wk = (const float*)d_in[6];
  const float* bk = (const float*)d_in[7];
  const float* Wv = (const float*)d_in[8];
  const float* bv = (const float*)d_in[9];
  const float* Wo = (const float*)d_in[10];
  const float* bo = (const float*)d_in[11];

  float* results = (float*)d_out;
  float* dist    = results + (size_t)NB * NQK * NF;

  const size_t elems = (size_t)NB * HEADS * NQK * DH;  // 4,194,304 bf16 each
  __bf16* qws = (__bf16*)d_ws;
  __bf16* kws = qws + elems;
  __bf16* vws = kws + elems;
  __bf16* cws = vws + elems;                            // 32 MB total

  const dim3 g(8, 32), t(256);
  gemm_kernel<float, true ><<<g, t, 0, stream>>>(queries, Wq, bq, (void*)qws, 0.125f); // 1/sqrt(64)
  gemm_kernel<float, true ><<<g, t, 0, stream>>>(keys,    Wk, bk, (void*)kws, 1.0f);
  gemm_kernel<float, true ><<<g, t, 0, stream>>>(values,  Wv, bv, (void*)vws, 1.0f);
  attn_kernel<<<dim3(64, 16), dim3(128), 0, stream>>>(qws, kws, vws, mask, dist, cws);
  gemm_kernel<__bf16, false><<<g, t, 0, stream>>>(cws, Wo, bo, (void*)results, 1.0f);
}
